// VointSelfAttention_19911468384717
// MI455X (gfx1250) — matrix operand order repaired and checked by reading, not compiled, not run on hardware
//
#include <hip/hip_runtime.h>
#include <stdint.h>

// ---------------------------------------------------------------------------
// VointSelfAttention for MI455X (gfx1250, wave32, WMMA f32_16x16x32_f16 + TDM)
// B=8, C=256, N=1024, V=12, H=4, D=64.  S_per_batch = N*V = 12288.
// ---------------------------------------------------------------------------

typedef __attribute__((ext_vector_type(16))) _Float16 v16h;
typedef __attribute__((ext_vector_type(8)))  _Float16 h8;
typedef __attribute__((ext_vector_type(8)))  float    v8f;
typedef __attribute__((ext_vector_type(4)))  unsigned int u32x4;
typedef __attribute__((ext_vector_type(8)))  int          i32x8;
typedef __attribute__((ext_vector_type(4)))  int          i32x4;

#define SPB   12288       // spatial elements per batch (N*V)
#define NB    8
#define NPTS  1024
#define NV    12
#define CCH   256
#define STOTF 98304.0f    // B*N*V (BN denominator)
#define LDA   40          // LDS row stride in halfs (80B = 64B row + 16B TDM pad)

#if defined(__has_builtin)
#if __has_builtin(__builtin_amdgcn_tensor_load_to_lds) && \
    __has_builtin(__builtin_amdgcn_s_wait_tensorcnt)
#define HAVE_TDM 1
#endif
#endif

// ---------------------------------------------------------------------------
// f32 -> f16 weight conversion (once per launch; enables TDM / b128 A staging)
// ---------------------------------------------------------------------------
__global__ __launch_bounds__(256) void k_cvt(const float* __restrict__ src,
                                             _Float16* __restrict__ dst, int n)
{
  int i = blockIdx.x * 256 + threadIdx.x;
  if (i < n) dst[i] = (_Float16)src[i];
}

// ---------------------------------------------------------------------------
// Fused 1x1-conv GEMM:  y[b,o,s] = (sum_c W[o,c]*act(in[b,c,s]) + bias[o]) * mask[b,s]
// act(v) = bnA ? (mask>0 ? relu(bnA[c]*v + bnB[c]) : 0) : v   (prev layer's BN fused)
// Accumulates per-channel sum/sumsq of the masked output into sums[] (atomics).
// Tiling: block=256 (8 waves). Tile M=64, N=128(spatial), K step 32.
// A tile staged by the Tensor Data Mover (f16 weights, LDS pad -> LDA stride).
// ---------------------------------------------------------------------------
__global__ __launch_bounds__(256) void k_gemm_conv(
    const float* __restrict__ in, const _Float16* __restrict__ Wh,
    const float* __restrict__ bias, const float* bnA, const float* bnB,
    const float* __restrict__ mask, float* __restrict__ y,
    float* __restrict__ sums, int Cin, int Cout)
{
  __shared__ __attribute__((aligned(16))) _Float16 As[64 * LDA];
  __shared__ __attribute__((aligned(16))) _Float16 Bs[128 * LDA];
  __shared__ float smask[128];

  const int tid  = threadIdx.x;
  const int lane = tid & 31;
  const int wave = tid >> 5;
  const int b    = blockIdx.z;
  const int o0   = blockIdx.y * 64;
  const int s0   = blockIdx.x * 128;

  if (tid < 128) smask[tid] = mask[(size_t)b * SPB + s0 + tid];
  __syncthreads();

  const int mblk = wave >> 1;     // 0..3
  const int nblk = wave & 1;      // 0..1
  const int half = lane >> 4;     // 0: lanes 0-15, 1: lanes 16-31
  const int l15  = lane & 15;

  v8f accs[4] = {};

  for (int k0 = 0; k0 < Cin; k0 += 32) {
#if defined(HAVE_TDM)
    // ---- A tile via TDM: 64x32 f16 tile of Wh -> LDS, 16B pad per 64B row
    if (wave == 0) {
      uint32_t lds_base = (uint32_t)(uintptr_t)(&As[0]);
      uint64_t ga = (uint64_t)(uintptr_t)(Wh + (size_t)o0 * Cin + k0);
      u32x4 g0;
      g0.x = 1u;                                   // count=1, user descriptor
      g0.y = lds_base;                             // lds_addr
      g0.z = (uint32_t)ga;                         // global_addr[31:0]
      g0.w = (uint32_t)(ga >> 32) | (2u << 30);    // global_addr[56:32], type=2
      i32x8 g1;
      g1.s0 = (1 << 16)      // data_size = 2 bytes
            | (1 << 20)      // pad_enable
            | (3 << 22)      // pad_interval: 16 DWORDs (= one 64B tile row)
            | (3 << 25);     // pad_amount: 4 DWORDs (16B) -> row stride 80B
      g1.s1 = (Cin & 0xFFFF) << 16;                          // tensor_dim0 lo
      g1.s2 = ((Cin >> 16) & 0xFFFF) | ((Cout & 0xFFFF) << 16); // dim0 hi, dim1 lo
      g1.s3 = ((Cout >> 16) & 0xFFFF) | (32 << 16);          // dim1 hi, tile_dim0=32
      g1.s4 = 64;                                            // tile_dim1=64, tile_dim2=0
      g1.s5 = Cin;                                           // tensor_dim0_stride lo
      g1.s6 = 0;                                             // stride0 hi, stride1 lo
      g1.s7 = 0;
      i32x4 gz4 = {0, 0, 0, 0};
      i32x8 gz8 = {0, 0, 0, 0, 0, 0, 0, 0};
      __builtin_amdgcn_tensor_load_to_lds(g0, g1, gz4, gz4, gz8, 0);
    }
#else
    // ---- A tile manually: one 16B copy per thread (64 rows x 4 h8 groups)
    {
      int row = tid >> 2, cg = tid & 3;
      *(h8*)&As[row * LDA + cg * 8] =
          *(const h8*)&Wh[(size_t)(o0 + row) * Cin + k0 + cg * 8];
    }
#endif

    // ---- B tile: act(in)[k0..k0+31][s0..s0+127] -> f16 LDS, [s][k] layout
    #pragma unroll
    for (int i = 0; i < 4; ++i) {
      int idx = tid + i * 256;               // 1024 float4 groups
      int cl = idx >> 5, sg = idx & 31;
      int sl = sg * 4;
      int c = k0 + cl;
      const float* src = &in[((size_t)b * Cin + c) * SPB + s0 + sl];
      if (k0 + 32 < Cin) __builtin_prefetch(src + 32 * SPB, 0, 0);
      float4 raw = *(const float4*)src;
      float vv[4] = {raw.x, raw.y, raw.z, raw.w};
      if (bnA) {
        float a = bnA[c], b2 = bnB[c];
        #pragma unroll
        for (int j = 0; j < 4; ++j) {
          float m = smask[sl + j];
          vv[j] = (m > 0.0f) ? fmaxf(0.0f, a * vv[j] + b2) : 0.0f;
        }
      }
      #pragma unroll
      for (int j = 0; j < 4; ++j)
        Bs[(sl + j) * LDA + cl] = (_Float16)vv[j];
    }

#if defined(HAVE_TDM)
    if (wave == 0) __builtin_amdgcn_s_wait_tensorcnt((short)0);
#endif
    __syncthreads();

    // ---- A fragment (ISA 16-bit A 16x32 layout):
    // lane<16: K {0..7,16..23}; lane>=16: K {8..15,24..31}; row m = l15.
    const int am  = mblk * 16 + l15;
    const int akb = half ? 8 : 0;
    h8 alo = *(const h8*)&As[am * LDA + akb];
    h8 ahi = *(const h8*)&As[am * LDA + akb + 16];
    v16h afrag = __builtin_shufflevector(alo, ahi,
        0,1,2,3,4,5,6,7,8,9,10,11,12,13,14,15);

    // ---- B fragments (col n = l15; lane<16: K {0..7,16..23}; lane>=16: K {8..15,24..31})
    const int bkb = half ? 8 : 0;
    #pragma unroll
    for (int sub = 0; sub < 4; ++sub) {
      int bn = nblk * 64 + sub * 16 + l15;
      h8 blo = *(const h8*)&Bs[bn * LDA + bkb];
      h8 bhi = *(const h8*)&Bs[bn * LDA + bkb + 16];
      v16h bfrag = __builtin_shufflevector(blo, bhi,
          0,1,2,3,4,5,6,7,8,9,10,11,12,13,14,15);
      accs[sub] = __builtin_amdgcn_wmma_f32_16x16x32_f16(
          false, afrag, false, bfrag, (short)0, accs[sub], false, false);
    }
    __syncthreads();
  }

  // ---- epilogue: bias, mask, store f32, per-channel sum/sumsq -> atomics
  float s1[8], s2[8];
  #pragma unroll
  for (int r = 0; r < 8; ++r) { s1[r] = 0.0f; s2[r] = 0.0f; }

  #pragma unroll
  for (int sub = 0; sub < 4; ++sub) {
    #pragma unroll
    for (int r = 0; r < 8; ++r) {
      int o  = o0 + mblk * 16 + (half << 3) + r;   // C/D layout: M = r + 8*(lane/16)
      int sl = nblk * 64 + sub * 16 + l15;         // N = lane&15
      float m   = smask[sl];
      float val = (accs[sub][r] + bias[o]) * m;
      y[((size_t)b * Cout + o) * SPB + s0 + sl] = val;
      s1[r] += val;
      s2[r] += val * val;
    }
  }
  #pragma unroll
  for (int r = 0; r < 8; ++r) {
    float a = s1[r], q = s2[r];
    a += __shfl_xor(a, 1, 32);  q += __shfl_xor(q, 1, 32);
    a += __shfl_xor(a, 2, 32);  q += __shfl_xor(q, 2, 32);
    a += __shfl_xor(a, 4, 32);  q += __shfl_xor(q, 4, 32);
    a += __shfl_xor(a, 8, 32);  q += __shfl_xor(q, 8, 32);
    if (l15 == 0) {
      int o = o0 + mblk * 16 + (half << 3) + r;
      atomicAdd(&sums[o], a);
      atomicAdd(&sums[Cout + o], q);
    }
  }
}

// ---------------------------------------------------------------------------
// BN stats -> per-channel affine:  a = g*rsqrt(var+eps), b = beta - mu*a
// ---------------------------------------------------------------------------
__global__ void k_finalize(const float* __restrict__ sums,
                           const float* __restrict__ g,
                           const float* __restrict__ bt,
                           float* __restrict__ bnA, float* __restrict__ bnB,
                           int Cout)
{
  int c = threadIdx.x;
  if (c < Cout) {
    float mu  = sums[c] / STOTF;
    float var = sums[Cout + c] / STOTF - mu * mu;
    float a   = g[c] * rsqrtf(var + 1e-5f);
    bnA[c] = a;
    bnB[c] = bt[c] - mu * a;
  }
}

// ---------------------------------------------------------------------------
// Per-(b,n) attention over V=12 views, H=4 heads, D=64.
// BN+ReLU+mask of qkv applied while staging to LDS. Reference's qbias is
// constant over the softmax axis -> plain softmax * query-mask; keys unmasked.
// ---------------------------------------------------------------------------
__global__ __launch_bounds__(256) void k_attn(
    const float* __restrict__ qkv, const float* __restrict__ bnA,
    const float* __restrict__ bnB, const float* __restrict__ mask,
    float* __restrict__ outp)
{
  __shared__ float sKQV[3 * 256 * 13];   // [part][ch][view], part 0=k,1=q,2=v
  __shared__ float sDots[4 * 12 * 12];
  __shared__ float sAttn[4 * 12 * 12];
  __shared__ float sm[12];

  const int tid = threadIdx.x;
  const int bn  = blockIdx.x;
  const int b   = bn / NPTS, n = bn % NPTS;

  if (tid < 12) sm[tid] = mask[(size_t)b * SPB + n * NV + tid];
  __syncthreads();

  #pragma unroll
  for (int i = 0; i < 36; ++i) {
    int e = tid + i * 256;                 // 3*256*12 = 9216 elements
    int part = e / 3072;
    int rem  = e - part * 3072;
    int ch = rem / 12, vv = rem % 12;
    int g = part * 256 + ch;
    float raw = qkv[((size_t)b * 768 + g) * SPB + n * NV + vv];
    float m = sm[vv];
    sKQV[part * 3328 + ch * 13 + vv] =
        (m > 0.0f) ? fmaxf(0.0f, bnA[g] * raw + bnB[g]) : 0.0f;
  }
  __syncthreads();

  // dots[h][vi][j] = (1/16) * sum_d q[h*64+d][vi] * k[h*64+d][j]
  for (int e = tid; e < 576; e += 256) {
    int h = e / 144, r = e - h * 144;
    int vi = r / 12, j = r - vi * 12;
    const float* qp = &sKQV[3328 + (h * 64) * 13];
    const float* kp = &sKQV[(h * 64) * 13];
    float acc = 0.0f;
    #pragma unroll 8
    for (int d = 0; d < 64; ++d)
      acc += qp[d * 13 + vi] * kp[d * 13 + j];
    sDots[h * 144 + vi * 12 + j] = acc * 0.0625f;  // C^-0.5 = 1/16
  }
  __syncthreads();

  if (tid < 48) {
    int h = tid / 12, vi = tid - h * 12;
    const float* row = &sDots[h * 144 + vi * 12];
    float mx = row[0];
    #pragma unroll
    for (int j = 1; j < 12; ++j) mx = fmaxf(mx, row[j]);
    float ex[12], s = 0.0f;
    #pragma unroll
    for (int j = 0; j < 12; ++j) { ex[j] = __expf(row[j] - mx); s += ex[j]; }
    float inv = sm[vi] / s;
    #pragma unroll
    for (int j = 0; j < 12; ++j) sAttn[h * 144 + vi * 12 + j] = ex[j] * inv;
  }
  __syncthreads();

  #pragma unroll
  for (int i = 0; i < 12; ++i) {
    int e = tid + i * 256;                 // 256ch * 12views = 3072
    int co = e / 12, vi = e - co * 12;
    int h = co >> 6, d = co & 63;
    const float* vp = &sKQV[2 * 3328 + (h * 64 + d) * 13];
    const float* ar = &sAttn[h * 144 + vi * 12];
    float acc = 0.0f;
    #pragma unroll
    for (int j = 0; j < 12; ++j) acc += ar[j] * vp[j];
    outp[((size_t)b * 256 + co) * SPB + n * NV + vi] = acc;
  }
}

// ---------------------------------------------------------------------------
// out = xin + (mask>0 ? relu(bnA[c]*y + bnB[c]) : 0)
// ---------------------------------------------------------------------------
__global__ __launch_bounds__(256) void k_residual(
    const float* __restrict__ xin, const float* __restrict__ yv,
    const float* __restrict__ bnA, const float* __restrict__ bnB,
    const float* __restrict__ mask, float* __restrict__ outp)
{
  size_t e = (size_t)blockIdx.x * 256 + threadIdx.x;   // total divides evenly
  int s = (int)(e % SPB);
  size_t t = e / SPB;
  int c = (int)(t % CCH);
  int b = (int)(t / CCH);
  float m  = mask[(size_t)b * SPB + s];
  float yc = yv[e];
  float ap = (m > 0.0f) ? fmaxf(0.0f, bnA[c] * yc + bnB[c]) : 0.0f;
  outp[e] = xin[e] + ap;
}

// ---------------------------------------------------------------------------
extern "C" void kernel_launch(void* const* d_in, const int* in_sizes, int n_in,
                              void* d_out, int out_size, void* d_ws, size_t ws_size,
                              hipStream_t stream)
{
  (void)in_sizes; (void)n_in; (void)out_size; (void)ws_size;
  const float* x    = (const float*)d_in[0];
  const float* mask = (const float*)d_in[1];
  const float* We   = (const float*)d_in[2];
  const float* be   = (const float*)d_in[3];
  const float* ge   = (const float*)d_in[4];
  const float* bne  = (const float*)d_in[5];
  const float* Wa   = (const float*)d_in[6];
  const float* ba   = (const float*)d_in[7];
  const float* ga   = (const float*)d_in[8];
  const float* bna  = (const float*)d_in[9];
  const float* Wo   = (const float*)d_in[10];
  const float* bo   = (const float*)d_in[11];
  const float* go   = (const float*)d_in[12];
  const float* bno  = (const float*)d_in[13];
  const float* Wf   = (const float*)d_in[14];
  const float* bf   = (const float*)d_in[15];
  const float* gf   = (const float*)d_in[16];
  const float* bnf  = (const float*)d_in[17];

  // workspace layout (floats)
  float* ws   = (float*)d_ws;
  float* y_e  = ws;                       //  8*64*12288  = 6291456
  float* y_a  = y_e  + 6291456;           //  8*768*12288 = 75497472
  float* y_cf = y_a  + 75497472;          //  attn_out, later reused for y_f
  float* y_o  = y_cf + 25165824;
  float* x1   = y_o  + 25165824;
  float* st   = x1   + 25165824;          // stats: sums (2688) + affine (2688)
  float* sums_e = st;
  float* sums_a = st + 128;
  float* sums_o = st + 1664;
  float* sums_f = st + 2176;
  float* ab  = st + 2688;
  float* a_e = ab;          float* b_e = ab + 64;
  float* a_a = ab + 128;    float* b_a = ab + 128 + 768;
  float* a_o = ab + 1664;   float* b_o = ab + 1664 + 256;
  float* a_f = ab + 2176;   float* b_f = ab + 2176 + 256;
  _Float16* wh   = (_Float16*)(st + 5376);  // f16 weights (16B-aligned region)
  _Float16* wh_e = wh;                      // 64*256   = 16384
  _Float16* wh_a = wh + 16384;              // 768*64   = 49152
  _Float16* wh_o = wh + 65536;              // 256*256  = 65536
  _Float16* wh_f = wh + 131072;             // 256*256  = 65536

  (void)hipMemsetAsync(st, 0, 2688 * sizeof(float), stream);

  dim3 blk(256);

  // weights -> f16 (enables TDM / b128 A staging)
  k_cvt<<<64,  blk, 0, stream>>>(We, wh_e, 16384);
  k_cvt<<<192, blk, 0, stream>>>(Wa, wh_a, 49152);
  k_cvt<<<256, blk, 0, stream>>>(Wo, wh_o, 65536);
  k_cvt<<<256, blk, 0, stream>>>(Wf, wh_f, 65536);

  // emb = vconv(x, We)  : Cin=256 -> Cout=64, no input BN
  k_gemm_conv<<<dim3(96, 1, NB), blk, 0, stream>>>(
      x, wh_e, be, nullptr, nullptr, mask, y_e, sums_e, 256, 64);
  k_finalize<<<1, 64, 0, stream>>>(sums_e, ge, bne, a_e, b_e, 64);

  // qkv = vconv(bn(emb), Wa) : Cin=64 -> Cout=768, BN of emb fused into load
  k_gemm_conv<<<dim3(96, 12, NB), blk, 0, stream>>>(
      y_e, wh_a, ba, a_e, b_e, mask, y_a, sums_a, 64, 768);
  k_finalize<<<1, 768, 0, stream>>>(sums_a, ga, bna, a_a, b_a, 768);

  // attention (BN of qkv fused into LDS staging)
  k_attn<<<NB * NPTS, blk, 0, stream>>>(y_a, a_a, b_a, mask, y_cf);

  // conv_o : Cin=256 -> Cout=256, raw attn output as input
  k_gemm_conv<<<dim3(96, 4, NB), blk, 0, stream>>>(
      y_cf, wh_o, bo, nullptr, nullptr, mask, y_o, sums_o, 256, 256);
  k_finalize<<<1, 256, 0, stream>>>(sums_o, go, bno, a_o, b_o, 256);

  // x1 = x + bn_relu_mask(y_o)
  k_residual<<<98304, blk, 0, stream>>>(x, y_o, a_o, b_o, mask, x1);

  // conv_f : Cin=256 -> Cout=256 on x1 (reuse y_cf as y_f)
  k_gemm_conv<<<dim3(96, 4, NB), blk, 0, stream>>>(
      x1, wh_f, bf, nullptr, nullptr, mask, y_cf, sums_f, 256, 256);
  k_finalize<<<1, 256, 0, stream>>>(sums_f, gf, bnf, a_f, b_f, 256);

  // out = x1 + bn_relu_mask(y_f)
  k_residual<<<98304, blk, 0, stream>>>(x1, y_cf, a_f, b_f, mask, (float*)d_out);
}
